// MambaBlock_62792421867544
// MI455X (gfx1250) — matrix operand order repaired and checked by reading, not compiled, not run on hardware
//
#include <hip/hip_runtime.h>
#include <hip/hip_bf16.h>
#include <math.h>

typedef _Float16 f16;
typedef __attribute__((ext_vector_type(16))) _Float16 v16h;
typedef __attribute__((ext_vector_type(8)))  _Float16 v8h;
typedef __attribute__((ext_vector_type(8)))  float    v8f;

#define DEV __device__ __forceinline__

constexpr int Bsz = 2, L = 2048, D = 512, DI = 1024, DS = 16, DTR = 32;
constexpr int MR  = Bsz * L;      // 4096 rows per direction
constexpr int TCH = 8;            // time chunks for the parallel scan
constexpr int TLEN = L / TCH;     // 256

// ---------------------------------------------------------------------------
// WMMA fragment helpers (CDNA5 16x16x32 f16 layout, ISA 7.12.2)
//  A (16x32, MxK): lane<16 -> halfs = A[m, k0..k0+7] ++ A[m, k0+16..k0+23]
//                  lane>=16 -> +8 on the K base.
//  B as row-major W (N x K): same K layout as A (lane>=16 -> +8 on the K base).
// ---------------------------------------------------------------------------
DEV v16h make_frag(const f16* p0, const f16* p1) {
  union { v16h v; v8h h[2]; } u;
  u.h[0] = *(const v8h*)p0;
  u.h[1] = *(const v8h*)p1;
  return u.v;
}

// ---------------------------------------------------------------------------
// Generic WMMA GEMM: C[m,n] = sum_k A[m,k] * W[n,k]  (A: Mx K, W: N x K, f16)
// Block = 256 threads = 8 waves; wave computes a 16(M) x 64(N) tile.
// grid.x = M/128, grid.y = N/64. K multiple of 32.
// ---------------------------------------------------------------------------
template <class Epi>
__global__ __launch_bounds__(256) void wmma_gemm_kernel(
    const f16* __restrict__ A, int lda,
    const f16* __restrict__ W, int ldw,
    int K, Epi epi) {
  const int lane = threadIdx.x & 31;
  const int wave = threadIdx.x >> 5;
  const int row  = blockIdx.x * 128 + wave * 16;
  const int col  = blockIdx.y * 64;
  const int mn   = lane & 15;
  const int ka   = (lane < 16) ? 0 : 8;
  const int kb   = (lane < 16) ? 0 : 8;
  const f16* ap  = A + (size_t)(row + mn) * lda + ka;
  const f16* wp0 = W + (size_t)(col + mn) * ldw + kb;
  const f16* wp1 = wp0 + (size_t)16 * ldw;
  const f16* wp2 = wp0 + (size_t)32 * ldw;
  const f16* wp3 = wp0 + (size_t)48 * ldw;

  v8f acc0 = {}, acc1 = {}, acc2 = {}, acc3 = {};
  for (int k0 = 0; k0 < K; k0 += 32) {
    v16h a  = make_frag(ap + k0, ap + k0 + 16);
    v16h b0 = make_frag(wp0 + k0, wp0 + k0 + 16);
    acc0 = __builtin_amdgcn_wmma_f32_16x16x32_f16(false, a, false, b0, (short)0, acc0, false, false);
    v16h b1 = make_frag(wp1 + k0, wp1 + k0 + 16);
    acc1 = __builtin_amdgcn_wmma_f32_16x16x32_f16(false, a, false, b1, (short)0, acc1, false, false);
    v16h b2 = make_frag(wp2 + k0, wp2 + k0 + 16);
    acc2 = __builtin_amdgcn_wmma_f32_16x16x32_f16(false, a, false, b2, (short)0, acc2, false, false);
    v16h b3 = make_frag(wp3 + k0, wp3 + k0 + 16);
    acc3 = __builtin_amdgcn_wmma_f32_16x16x32_f16(false, a, false, b3, (short)0, acc3, false, false);
  }
  const int moff = (lane < 16) ? 0 : 8;   // C/D layout: VGPR r -> M = moff + r
  v8f accs[4] = {acc0, acc1, acc2, acc3};
#pragma unroll
  for (int t = 0; t < 4; ++t)
#pragma unroll
    for (int r = 0; r < 8; ++r)
      epi(row + moff + r, col + t * 16 + mn, accs[t][r]);
}

// ---- epilogues -------------------------------------------------------------
struct EpiF32 {        // plain f32 store
  float* C; int ldc;
  __device__ void operator()(int m, int n, float v) const { C[(size_t)m * ldc + n] = v; }
};
struct EpiHalfOff {    // f16 store with column offset (merged out/comb weight)
  f16* C; int ldc; int coloff;
  __device__ void operator()(int m, int n, float v) const { C[(size_t)m * ldc + coloff + n] = (f16)v; }
};
struct EpiXproj {      // dbc: full row f32 (B,C live here); dt-slice also f16 for next GEMM
  float* dbc; f16* dbcdt;
  __device__ void operator()(int m, int n, float v) const {
    dbc[(size_t)m * 64 + n] = v;
    if (n < DTR) dbcdt[(size_t)m * DTR + n] = (f16)v;
  }
};
struct EpiDt {         // dt = softplus(acc + dt_b[n])
  const float* bias; float* dt;
  __device__ void operator()(int m, int n, float v) const {
    float t = v + bias[n];
    dt[(size_t)m * DI + n] = (t > 20.f) ? t : log1pf(__expf(t));
  }
};
struct EpiFinal {      // out = acc + comb_b[n] + x[m,n]   (residual)
  const float* x; const float* cb; float* out;
  __device__ void operator()(int m, int n, float v) const {
    out[(size_t)m * D + n] = v + cb[n] + x[(size_t)m * D + n];
  }
};

// ---------------------------------------------------------------------------
// LayerNorm: one block per row of 512, emits f16 operand for the in-proj GEMM.
// ---------------------------------------------------------------------------
__global__ __launch_bounds__(256) void layernorm_kernel(
    const float* __restrict__ x, const float* __restrict__ g,
    const float* __restrict__ b, f16* __restrict__ xn) {
  const int row = blockIdx.x;
  const float* xr = x + (size_t)row * D;
  float v0 = xr[threadIdx.x], v1 = xr[threadIdx.x + 256];
  float s = v0 + v1, q = v0 * v0 + v1 * v1;
#pragma unroll
  for (int off = 16; off > 0; off >>= 1) {
    s += __shfl_down(s, off, 32);
    q += __shfl_down(q, off, 32);
  }
  __shared__ float ls[8], lq[8], stat[2];
  const int lane = threadIdx.x & 31, w = threadIdx.x >> 5;
  if (lane == 0) { ls[w] = s; lq[w] = q; }
  __syncthreads();
  if (threadIdx.x == 0) {
    float S = 0.f, Q = 0.f;
#pragma unroll
    for (int i = 0; i < 8; ++i) { S += ls[i]; Q += lq[i]; }
    stat[0] = S; stat[1] = Q;
  }
  __syncthreads();
  const float mean = stat[0] * (1.f / D);
  const float var  = stat[1] * (1.f / D) - mean * mean;
  const float rstd = rsqrtf(var + 1e-5f);
  xn[(size_t)row * D + threadIdx.x]       = (f16)((v0 - mean) * rstd * g[threadIdx.x]       + b[threadIdx.x]);
  xn[(size_t)row * D + threadIdx.x + 256] = (f16)((v1 - mean) * rstd * g[threadIdx.x + 256] + b[threadIdx.x + 256]);
}

// ---------------------------------------------------------------------------
// Causal depthwise conv (dc=4) + SiLU, with time-index map for backward dir.
// Emits xc in scan order, both f32 (for scan) and f16 (for xproj GEMM).
// ---------------------------------------------------------------------------
__global__ __launch_bounds__(256) void conv_silu_kernel(
    const float* __restrict__ xz_f, const float* __restrict__ xz_b,
    const float* __restrict__ fw, const float* __restrict__ fb,
    const float* __restrict__ bw, const float* __restrict__ bb,
    float* __restrict__ xc32, f16* __restrict__ xc16) {
  const int gid = blockIdx.x * 256 + threadIdx.x;
  const int d   = gid & (DI - 1);
  const int s   = (gid >> 10) & (L - 1);
  const int b   = (gid >> 21) & 1;
  const int dir = gid >> 22;
  const float* xz = dir ? xz_b : xz_f;
  const float* cw = dir ? bw : fw;
  const float* cb = dir ? bb : fb;
  float acc = cb[d];
#pragma unroll
  for (int k = 0; k < 4; ++k) {
    const int sp = s - 3 + k;
    if (sp >= 0) {
      const int t = dir ? (L - 1 - sp) : sp;
      acc += cw[d * 4 + k] * xz[((size_t)(b * L + t)) * (2 * DI) + d];
    }
  }
  const float sv = acc / (1.f + __expf(-acc));
  const size_t o = ((size_t)((dir * Bsz + b) * L + s)) * DI + d;
  xc32[o] = sv;
  xc16[o] = (f16)sv;
}

// ---------------------------------------------------------------------------
// Chunk-parallel selective scan. Linear recurrence h <- e^{dt*a} h + dt*xc*B,
// diagonal transition => chunk product = e^{a * sum(dt)}.
//  Phase A (FINALPASS=false): 8 chunks in parallel, zero-init, record sum(dt)
//  and chunk-final h.  Phase B: tiny sequential combine.  Phase C: re-scan
//  with corrected initial h, emit g = (y + D*xc) * silu(z) in f16 (time order).
// ---------------------------------------------------------------------------
template <bool FINALPASS>
__global__ __launch_bounds__(256) void scan_kernel(
    const float* __restrict__ dt, const float* __restrict__ xc,
    const float* __restrict__ dbc,
    const float* __restrict__ xz_f, const float* __restrict__ xz_b,
    const float* __restrict__ A_f, const float* __restrict__ A_b,
    const float* __restrict__ D_f, const float* __restrict__ D_b,
    const float* __restrict__ hinit, float* __restrict__ hfin,
    float* __restrict__ sumdt, f16* __restrict__ g) {
  const int blk = blockIdx.x;             // 128 blocks: dir(2) x b(2) x dch(4) x j(8)
  const int j   = blk & 7;
  const int dch = (blk >> 3) & 3;
  const int b   = (blk >> 5) & 1;
  const int dir = blk >> 6;
  const int d   = dch * 256 + threadIdx.x;
  const int db  = dir * Bsz + b;
  const size_t rbase = (size_t)db * L;
  const float* dtp = dt + rbase * DI;
  const float* xcp = xc + rbase * DI;
  const float* bcp = dbc + rbase * 64;
  const float* Al  = (dir ? A_b : A_f) + (size_t)d * DS;
  float a[DS], h[DS];
#pragma unroll
  for (int n = 0; n < DS; ++n) a[n] = -__expf(Al[n]);
  if (FINALPASS) {
    const float* hi = hinit + ((size_t)(db * TCH + j)) * DI * DS + (size_t)d * DS;
#pragma unroll
    for (int n = 0; n < DS; ++n) h[n] = hi[n];
  } else {
#pragma unroll
    for (int n = 0; n < DS; ++n) h[n] = 0.f;
  }
  const float Dv = (dir ? D_b : D_f)[d];
  const float* xz = dir ? xz_b : xz_f;
  float sdt = 0.f;
  __shared__ float lbc[64][32];           // B (cols 0..15) and C (16..31) per t
  for (int sub = 0; sub < TLEN / 64; ++sub) {
    const int t0 = j * TLEN + sub * 64;
    __syncthreads();
    for (int i = threadIdx.x; i < 64 * 32; i += 256) {
      const int r = i >> 5, cc = i & 31;
      lbc[r][cc] = bcp[(size_t)(t0 + r) * 64 + 32 + cc];
    }
    __syncthreads();
    for (int tt = 0; tt < 64; ++tt) {
      const int t = t0 + tt;
      const float dtv = dtp[(size_t)t * DI + d];
      const float xcv = xcp[(size_t)t * DI + d];
      const float dx  = dtv * xcv;
      if (!FINALPASS) sdt += dtv;
      float y = 0.f;
#pragma unroll
      for (int n = 0; n < DS; ++n) {
        h[n] = __expf(dtv * a[n]) * h[n] + dx * lbc[tt][n];
        y += h[n] * lbc[tt][16 + n];
      }
      if (FINALPASS) {
        y += Dv * xcv;
        const int tw = dir ? (L - 1 - t) : t;
        const float zv = xz[((size_t)(b * L + tw)) * (2 * DI) + DI + d];
        const float sz = zv / (1.f + __expf(-zv));
        g[((size_t)(b * L + tw)) * (2 * DI) + dir * DI + d] = (f16)(y * sz);
      }
    }
  }
  if (!FINALPASS) {
    float* hf = hfin + ((size_t)(db * TCH + j)) * DI * DS + (size_t)d * DS;
#pragma unroll
    for (int n = 0; n < DS; ++n) hf[n] = h[n];
    sumdt[(size_t)(db * TCH + j) * DI + d] = sdt;
  }
}

__global__ __launch_bounds__(256) void scan_combine_kernel(
    const float* __restrict__ A_f, const float* __restrict__ A_b,
    const float* __restrict__ hfin, const float* __restrict__ sumdt,
    float* __restrict__ hinit) {
  const int gid = blockIdx.x * 256 + threadIdx.x;   // 65536 = 4*1024*16
  const int n   = gid & 15;
  const int d   = (gid >> 4) & (DI - 1);
  const int db  = gid >> 14;
  const int dir = db >> 1;
  const float a = -__expf((dir ? A_b : A_f)[(size_t)d * DS + n]);
  float acc = 0.f;
  for (int j = 0; j < TCH; ++j) {
    const size_t idx = ((size_t)(db * TCH + j)) * DI * DS + (size_t)d * DS + n;
    hinit[idx] = acc;
    acc = __expf(sumdt[(size_t)(db * TCH + j) * DI + d] * a) * acc + hfin[idx];
  }
}

// ---------------------------------------------------------------------------
// Small weight-prep kernels (f32 -> f16 operand staging)
// ---------------------------------------------------------------------------
__global__ __launch_bounds__(256) void cast_f16_kernel(const float* __restrict__ s,
                                                       f16* __restrict__ d, int n) {
  const int i = blockIdx.x * 256 + threadIdx.x;
  if (i < n) d[i] = (f16)s[i];
}
__global__ __launch_bounds__(256) void slice_cast_kernel(const float* __restrict__ s,
    f16* __restrict__ d, int cols, int srcld, int coloff, int n) {
  const int i = blockIdx.x * 256 + threadIdx.x;
  if (i < n) {
    const int r = i / cols, c = i - r * cols;
    d[(size_t)r * cols + c] = (f16)s[(size_t)r * srcld + coloff + c];
  }
}
__global__ __launch_bounds__(256) void transpose_cast_kernel(const float* __restrict__ s,
    f16* __restrict__ d, int R, int C) {  // s: RxC, d: CxR
  const int i = blockIdx.x * 256 + threadIdx.x;
  if (i < R * C) {
    const int r = i / C, c = i - r * C;
    d[(size_t)c * R + r] = (f16)s[i];
  }
}

// ---------------------------------------------------------------------------
extern "C" void kernel_launch(void* const* d_in, const int* in_sizes, int n_in,
                              void* d_out, int out_size, void* d_ws, size_t ws_size,
                              hipStream_t stream) {
  (void)in_sizes; (void)n_in; (void)out_size; (void)ws_size;
  // setup_inputs() dict order:
  const float* x       = (const float*)d_in[0];
  const float* ln_g    = (const float*)d_in[1];
  const float* ln_b    = (const float*)d_in[2];
  const float* comb_w  = (const float*)d_in[3];
  const float* comb_b  = (const float*)d_in[4];
  const float* in_w[2]    = {(const float*)d_in[5],  (const float*)d_in[14]};
  const float* conv_w[2]  = {(const float*)d_in[6],  (const float*)d_in[15]};
  const float* conv_b[2]  = {(const float*)d_in[7],  (const float*)d_in[16]};
  const float* xproj_w[2] = {(const float*)d_in[8],  (const float*)d_in[17]};
  const float* dt_w[2]    = {(const float*)d_in[9],  (const float*)d_in[18]};
  const float* dt_b[2]    = {(const float*)d_in[10], (const float*)d_in[19]};
  const float* A_log[2]   = {(const float*)d_in[11], (const float*)d_in[20]};
  const float* Dp[2]      = {(const float*)d_in[12], (const float*)d_in[21]};
  const float* out_w[2]   = {(const float*)d_in[13], (const float*)d_in[22]};
  float* out = (float*)d_out;

  // ---- workspace carve-out (256B aligned) ----
  char* ws = (char*)d_ws;
  size_t off = 0;
  auto carve = [&](size_t bytes) -> void* {
    off = (off + 255) & ~(size_t)255;
    void* p = ws + off;
    off += bytes;
    return p;
  };
  f16*   xn16   = (f16*)carve((size_t)MR * D * 2);                  // 4 MB
  float* xz[2]  = {(float*)carve((size_t)MR * 2 * DI * 4),          // 32 MB each
                   (float*)carve((size_t)MR * 2 * DI * 4)};
  float* xc32   = (float*)carve((size_t)2 * MR * DI * 4);           // 32 MB
  f16*   xc16   = (f16*)carve((size_t)2 * MR * DI * 2);             // 16 MB
  float* dbc    = (float*)carve((size_t)2 * MR * 64 * 4);           // 2 MB
  f16*   dbcdt  = (f16*)carve((size_t)2 * MR * DTR * 2);            // 0.5 MB
  float* dt32   = (float*)carve((size_t)2 * MR * DI * 4);           // 32 MB
  f16*   g16    = (f16*)carve((size_t)MR * 2 * DI * 2);             // 16 MB
  float* hfin   = (float*)carve((size_t)4 * TCH * DI * DS * 4);     // 2 MB
  float* hinit  = (float*)carve((size_t)4 * TCH * DI * DS * 4);     // 2 MB
  float* sumdt  = (float*)carve((size_t)4 * TCH * DI * 4);          // 128 KB
  f16*   w_in16[2]  = {(f16*)carve((size_t)2 * DI * D * 2), (f16*)carve((size_t)2 * DI * D * 2)};
  f16*   w_xp16[2]  = {(f16*)carve((size_t)64 * DI * 2),    (f16*)carve((size_t)64 * DI * 2)};
  f16*   w_dt16[2]  = {(f16*)carve((size_t)DI * DTR * 2),   (f16*)carve((size_t)DI * DTR * 2)};
  f16*   combA[2]   = {(f16*)carve((size_t)D * D * 2),      (f16*)carve((size_t)D * D * 2)};
  f16*   owT[2]     = {(f16*)carve((size_t)DI * D * 2),     (f16*)carve((size_t)DI * D * 2)};
  f16*   wm16       = (f16*)carve((size_t)D * 2 * DI * 2);          // 2 MB

  // ---- 1) weight staging to f16 ----
  for (int dir = 0; dir < 2; ++dir) {
    cast_f16_kernel<<<(2 * DI * D + 255) / 256, 256, 0, stream>>>(in_w[dir], w_in16[dir], 2 * DI * D);
    cast_f16_kernel<<<(64 * DI + 255) / 256, 256, 0, stream>>>(xproj_w[dir], w_xp16[dir], 64 * DI);
    cast_f16_kernel<<<(DI * DTR + 255) / 256, 256, 0, stream>>>(dt_w[dir], w_dt16[dir], DI * DTR);
    slice_cast_kernel<<<(D * D + 255) / 256, 256, 0, stream>>>(comb_w, combA[dir], D, 2 * D, dir * D, D * D);
    transpose_cast_kernel<<<(D * DI + 255) / 256, 256, 0, stream>>>(out_w[dir], owT[dir], D, DI);
  }
  // merged tail weight  Wm[:, dir*DI + k] = comb_w[:, dir*D:] @ out_w_dir  (WMMA)
  for (int dir = 0; dir < 2; ++dir) {
    wmma_gemm_kernel<EpiHalfOff><<<dim3(D / 128, DI / 64), 256, 0, stream>>>(
        combA[dir], D, owT[dir], D, D, EpiHalfOff{wm16, 2 * DI, dir * DI});
  }

  // ---- 2) LayerNorm ----
  layernorm_kernel<<<MR, 256, 0, stream>>>(x, ln_g, ln_b, xn16);

  // ---- 3) in-proj GEMMs (both directions, flip-free) ----
  for (int dir = 0; dir < 2; ++dir) {
    wmma_gemm_kernel<EpiF32><<<dim3(MR / 128, (2 * DI) / 64), 256, 0, stream>>>(
        xn16, D, w_in16[dir], D, D, EpiF32{xz[dir], 2 * DI});
  }

  // ---- 4) causal conv + SiLU (scan order) ----
  conv_silu_kernel<<<(2 * MR * DI) / 256, 256, 0, stream>>>(
      xz[0], xz[1], conv_w[0], conv_b[0], conv_w[1], conv_b[1], xc32, xc16);

  // ---- 5) x-proj and dt GEMMs ----
  for (int dir = 0; dir < 2; ++dir) {
    wmma_gemm_kernel<EpiXproj><<<dim3(MR / 128, 1), 256, 0, stream>>>(
        xc16 + (size_t)dir * MR * DI, DI, w_xp16[dir], DI, DI,
        EpiXproj{dbc + (size_t)dir * MR * 64, dbcdt + (size_t)dir * MR * DTR});
  }
  for (int dir = 0; dir < 2; ++dir) {
    wmma_gemm_kernel<EpiDt><<<dim3(MR / 128, DI / 64), 256, 0, stream>>>(
        dbcdt + (size_t)dir * MR * DTR, DTR, w_dt16[dir], DTR, DTR,
        EpiDt{dt_b[dir], dt32 + (size_t)dir * MR * DI});
  }

  // ---- 6) chunk-parallel selective scan (A -> combine -> C) ----
  scan_kernel<false><<<128, 256, 0, stream>>>(
      dt32, xc32, dbc, xz[0], xz[1], A_log[0], A_log[1], Dp[0], Dp[1],
      nullptr, hfin, sumdt, nullptr);
  scan_combine_kernel<<<256, 256, 0, stream>>>(A_log[0], A_log[1], hfin, sumdt, hinit);
  scan_kernel<true><<<128, 256, 0, stream>>>(
      dt32, xc32, dbc, xz[0], xz[1], A_log[0], A_log[1], Dp[0], Dp[1],
      hinit, nullptr, nullptr, g16);

  // ---- 7) fused out-proj + comb + residual ----
  wmma_gemm_kernel<EpiFinal><<<dim3(MR / 128, D / 64), 256, 0, stream>>>(
      g16, 2 * DI, wm16, 2 * DI, 2 * DI, EpiFinal{x, comb_b, out});
}
